// SelfVAE_50182397886998
// MI455X (gfx1250) — matrix operand order repaired and checked by reading, not compiled, not run on hardware
//
#include <hip/hip_runtime.h>
#include <hip/hip_bf16.h>
#include <cmath>

typedef __attribute__((ext_vector_type(16))) _Float16 v16h;
typedef __attribute__((ext_vector_type(8)))  _Float16 v8h;
typedef __attribute__((ext_vector_type(8)))  float    v8f;

#define DEV __device__ __forceinline__

// ---------------------------------------------------------------------------
// WMMA operand loaders (per CDNA5 ISA 7.12.2 VGPR layouts, wave32)
// A (16x32 f16, row-major [M,K]): lane l -> row m = l&15;
//   elems[0..7]  = K kb..kb+7    (kb = (l<16)?0:8)
//   elems[8..15] = K kb+16..kb+23
// B (32x16 f16 = W^T with W row-major [N,K]): lane l -> col n = l&15;
//   elems[0..7] = K kb..kb+7, elems[8..15] = K kb+16..kb+23   (kb = (l<16)?0:8)
// ---------------------------------------------------------------------------
DEV v16h load_a_tile(const _Float16* __restrict__ base, int ld, int lane) {
  int r  = lane & 15;
  int kb = (lane & 16) ? 8 : 0;
  const _Float16* p = base + (size_t)r * ld + kb;
  v8h lo = *(const v8h*)(p);
  v8h hi = *(const v8h*)(p + 16);
  v16h o;
#pragma unroll
  for (int i = 0; i < 8; ++i) { o[i] = lo[i]; o[i + 8] = hi[i]; }
  return o;
}

DEV v16h load_b_tile(const _Float16* __restrict__ base, int ld, int lane) {
  int n  = lane & 15;
  int kb = (lane & 16) ? 8 : 0;
  const _Float16* p = base + (size_t)n * ld + kb;
  v8h lo = *(const v8h*)(p);
  v8h hi = *(const v8h*)(p + 16);
  v16h o;
#pragma unroll
  for (int i = 0; i < 8; ++i) { o[i] = lo[i]; o[i + 8] = hi[i]; }
  return o;
}

DEV float sigmoidf_(float x) { return 1.0f / (1.0f + __expf(-x)); }

// ---------------------------------------------------------------------------
// Generic WMMA GEMM: C[M,N](f32) = A[M,K](f16) @ W[N,K]^T(f16) + bias + bias2
// One wave per 16x16 output tile. Optional f16 copy of C.
// M,N multiples of 16; K multiple of 32.
// ---------------------------------------------------------------------------
__global__ void wmma_gemm_kernel(const _Float16* __restrict__ A,
                                 const _Float16* __restrict__ W,
                                 const float* __restrict__ bias,
                                 const float* __restrict__ bias2,
                                 float* __restrict__ C,
                                 _Float16* __restrict__ Ch,
                                 int M, int N, int K) {
  int lane = threadIdx.x & 31;
  int wave = threadIdx.x >> 5;
  int tile = blockIdx.x * (blockDim.x >> 5) + wave;
  int tilesN = N >> 4;
  int tilesM = M >> 4;
  if (tile >= tilesM * tilesN) return;     // wave-uniform: EXEC stays all-ones
  int tm = tile / tilesN;
  int tn = tile - tm * tilesN;
  const _Float16* Ab = A + (size_t)tm * 16 * K;
  const _Float16* Wb = W + (size_t)tn * 16 * K;
  v8f acc = {};
  for (int k = 0; k < K; k += 32) {
    if (k + 256 < K) {                     // gfx1250 global_prefetch_b8
      __builtin_prefetch(Ab + k + 256, 0, 1);
      __builtin_prefetch(Wb + k + 256, 0, 1);
    }
    v16h a = load_a_tile(Ab + k, K, lane);
    v16h b = load_b_tile(Wb + k, K, lane);
    acc = __builtin_amdgcn_wmma_f32_16x16x32_f16(false, a, false, b,
                                                 (short)0, acc, false, false);
  }
  int n  = tn * 16 + (lane & 15);
  float bv = 0.f;
  if (bias)  bv += bias[n];
  if (bias2) bv += bias2[n];
  int m0 = tm * 16 + ((lane & 16) ? 8 : 0);
#pragma unroll
  for (int r = 0; r < 8; ++r) {
    float v = acc[r] + bv;
    size_t idx = (size_t)(m0 + r) * N + n;
    if (C)  C[idx]  = v;
    if (Ch) Ch[idx] = (_Float16)v;
  }
}

// ---------------------------------------------------------------------------
// Fused LSTM step. gates = (X @ Wih^T if X) + (IG if IG) + Hin @ Whh^T + biases
// Each wave owns a (16 batch x 16 hidden) tile and all 4 gates for it
// (A operand reused across 4 v_wmma issues), then applies the cell.
// B=64, H=1024 fixed -> 256 waves -> grid 32 x 256 threads.
// ---------------------------------------------------------------------------
__global__ void lstm_step_kernel(const _Float16* __restrict__ X, int Kx,
                                 const _Float16* __restrict__ Wih,
                                 const float* __restrict__ IG, int igStride,
                                 const _Float16* __restrict__ Hin,
                                 const _Float16* __restrict__ Whh,
                                 const float* __restrict__ bih,
                                 const float* __restrict__ bhh,
                                 float* __restrict__ Cst,
                                 _Float16* __restrict__ HoutH,
                                 float* __restrict__ Hstore32, int hs32Stride,
                                 _Float16* __restrict__ Hstore16, int hs16Stride) {
  const int H = 1024;
  int lane = threadIdx.x & 31;
  int wave = threadIdx.x >> 5;
  int tile = blockIdx.x * (blockDim.x >> 5) + wave;  // 0..255
  int tm = tile >> 6;   // batch tile 0..3
  int tj = tile & 63;   // hidden tile 0..63
  v8f z = {};
  v8f acc[4] = {z, z, z, z};
  if (X) {
    const _Float16* Ab = X + (size_t)tm * 16 * Kx;
    for (int k = 0; k < Kx; k += 32) {
      v16h a = load_a_tile(Ab + k, Kx, lane);
#pragma unroll
      for (int g = 0; g < 4; ++g) {
        const _Float16* Wb = Wih + ((size_t)(g * H + tj * 16)) * Kx + k;
        v16h b = load_b_tile(Wb, Kx, lane);
        acc[g] = __builtin_amdgcn_wmma_f32_16x16x32_f16(false, a, false, b,
                                                        (short)0, acc[g], false, false);
      }
    }
  }
  {
    const _Float16* Ab = Hin + (size_t)tm * 16 * H;
    for (int k = 0; k < H; k += 32) {
      v16h a = load_a_tile(Ab + k, H, lane);
#pragma unroll
      for (int g = 0; g < 4; ++g) {
        const _Float16* Wb = Whh + ((size_t)(g * H + tj * 16)) * H + k;
        v16h b = load_b_tile(Wb, H, lane);
        acc[g] = __builtin_amdgcn_wmma_f32_16x16x32_f16(false, a, false, b,
                                                        (short)0, acc[g], false, false);
      }
    }
  }
  int col = tj * 16 + (lane & 15);
  int m0  = tm * 16 + ((lane & 16) ? 8 : 0);
  float bi[4] = {0.f, 0.f, 0.f, 0.f};
#pragma unroll
  for (int g = 0; g < 4; ++g) {
    int n = g * H + col;
    if (bih) bi[g] += bih[n];
    if (bhh) bi[g] += bhh[n];
  }
#pragma unroll
  for (int r = 0; r < 8; ++r) {
    int b = m0 + r;
    float gi = acc[0][r] + bi[0];
    float gf = acc[1][r] + bi[1];
    float gg = acc[2][r] + bi[2];
    float go = acc[3][r] + bi[3];
    if (IG) {
      const float* ig = IG + (size_t)b * igStride;
      gi += ig[col];
      gf += ig[H + col];
      gg += ig[2 * H + col];
      go += ig[3 * H + col];
    }
    size_t ci = (size_t)b * H + col;
    float c_old = Cst[ci];
    float cn = sigmoidf_(gf) * c_old + sigmoidf_(gi) * tanhf(gg);
    float hn = sigmoidf_(go) * tanhf(cn);
    Cst[ci] = cn;
    HoutH[ci] = (_Float16)hn;
    if (Hstore32) Hstore32[(size_t)b * hs32Stride + col] = hn;
    if (Hstore16) Hstore16[(size_t)b * hs16Stride + col] = (_Float16)hn;
  }
}

// ---------------------------------------------------------------------------
// Small helper kernels
// ---------------------------------------------------------------------------
__global__ void cvt_f32_f16_kernel(const float* __restrict__ s,
                                   _Float16* __restrict__ d, long n) {
  long i = (long)blockIdx.x * blockDim.x + threadIdx.x;
  if (i < n) d[i] = (_Float16)s[i];
}

__global__ void cvt_rows_kernel(const float* __restrict__ src, int srcLd,
                                _Float16* __restrict__ dst, int w, int rows) {
  int i = blockIdx.x * blockDim.x + threadIdx.x;
  if (i >= rows * w) return;
  int r = i / w, c = i - r * w;
  dst[i] = (_Float16)src[(size_t)r * srcLd + c];
}

// out[(t*Bn+b)*E + e] = emb[idx[b*Tn+t]*E + e]  (time-major rows for step GEMMs)
__global__ void gather_emb_kernel(const float* __restrict__ emb,
                                  const int* __restrict__ idx,
                                  _Float16* __restrict__ out,
                                  int Tn, int Bn, int E_) {
  long i = (long)blockIdx.x * blockDim.x + threadIdx.x;
  long total = (long)Tn * Bn * E_;
  if (i >= total) return;
  int e = (int)(i % E_);
  long row = i / E_;
  int b = (int)(row % Bn);
  int t = (int)(row / Bn);
  int tok = idx[b * Tn + t];
  out[i] = (_Float16)emb[(long)tok * E_ + e];
}

__global__ void concat4_kernel(const _Float16* __restrict__ s0, int w0,
                               const _Float16* __restrict__ s1, int w1,
                               const _Float16* __restrict__ s2, int w2,
                               const _Float16* __restrict__ s3, int w3,
                               _Float16* __restrict__ out, int rows) {
  int W = w0 + w1 + w2 + w3;
  int i = blockIdx.x * blockDim.x + threadIdx.x;
  if (i >= rows * W) return;
  int r = i / W, c = i - r * W;
  _Float16 v;
  if (c < w0)                 v = s0[(size_t)r * w0 + c];
  else if (c < w0 + w1)       v = s1[(size_t)r * w1 + (c - w0)];
  else if (c < w0 + w1 + w2)  v = s2[(size_t)r * w2 + (c - w0 - w1)];
  else                        v = s3[(size_t)r * w3 + (c - w0 - w1 - w2)];
  out[i] = v;
}

// masked softmax over T=64 + context = attn_w @ enc_output  (one block per batch)
__global__ void attn_context_kernel(const float* __restrict__ scale,
                                    const int* __restrict__ enc_in,
                                    const float* __restrict__ encOut,  // [B,T,H]
                                    float* __restrict__ ctx32,
                                    _Float16* __restrict__ ctx16) {
  const int T = 64, H = 1024, PAD = 31999;
  int b = blockIdx.x;
  int tid = threadIdx.x;
  __shared__ float w[64];
  __shared__ float sinv;
  if (tid < T) {
    float s = scale[b * T + tid];
    if (enc_in[b * T + tid] == PAD) s = -INFINITY;
    w[tid] = s;
  }
  __syncthreads();
  if (tid == 0) {
    float mx = -INFINITY;
    for (int t = 0; t < T; ++t) mx = fmaxf(mx, w[t]);
    float sum = 0.f;
    for (int t = 0; t < T; ++t) { float e = __expf(w[t] - mx); w[t] = e; sum += e; }
    sinv = 1.f / sum;
  }
  __syncthreads();
  float inv = sinv;
  for (int h0 = tid; h0 < H; h0 += blockDim.x) {
    float s = 0.f;
    for (int t = 0; t < T; ++t) s += w[t] * encOut[((size_t)b * T + t) * H + h0];
    s *= inv;
    ctx32[(size_t)b * H + h0] = s;
    ctx16[(size_t)b * H + h0] = (_Float16)s;
  }
}

__global__ void kld_kernel(const float* __restrict__ pri,
                           const float* __restrict__ inf_,
                           const int* __restrict__ length, int t,
                           float* __restrict__ out_kld) {
  int b = blockIdx.x;
  int zi = threadIdx.x;  // 128
  float pm  = pri[b * 256 + zi],  plv = pri[b * 256 + 128 + zi];
  float im  = inf_[b * 256 + zi], ilv = inf_[b * 256 + 128 + zi];
  float d = im - pm;
  float term = plv - ilv + __expf(ilv - plv) + d * d / __expf(plv) - 1.0f;
  __shared__ float s[128];
  s[zi] = term;
  __syncthreads();
  for (int o = 64; o > 0; o >>= 1) { if (zi < o) s[zi] += s[zi + o]; __syncthreads(); }
  if (zi == 0) {
    float mask = (length[b] + 1 > t) ? 1.f : 0.f;
    atomicAdd(out_kld, 0.5f * s[0] * mask);
  }
}

__global__ void auxc_kernel(const float* __restrict__ ctx,
                            const float* __restrict__ auxMu,
                            float* __restrict__ out_aux) {
  int b = blockIdx.x;
  int tid = threadIdx.x;  // 256
  float s = 0.f;
  for (int j = tid; j < 1024; j += 256) {
    float d = ctx[(size_t)b * 1024 + j] - auxMu[(size_t)b * 1024 + j];
    s += d * d;
  }
  __shared__ float r[256];
  r[tid] = s;
  __syncthreads();
  for (int o = 128; o > 0; o >>= 1) { if (tid < o) r[tid] += r[tid + o]; __syncthreads(); }
  if (tid == 0) atomicAdd(out_aux, r[0]);
}

__global__ void latent_z_kernel(const float* __restrict__ lat,
                                const float* __restrict__ eps,
                                float* __restrict__ mu_out,
                                float* __restrict__ lv_out,
                                _Float16* __restrict__ z16) {
  int i = blockIdx.x * blockDim.x + threadIdx.x;  // B*Z = 8192
  if (i >= 64 * 128) return;
  int b = i >> 7, zz = i & 127;
  float mu = lat[b * 256 + zz], lv = lat[b * 256 + 128 + zz];
  float zv = mu + __expf(0.5f * lv) * eps[i];
  mu_out[i] = mu;
  lv_out[i] = lv;
  z16[i] = (_Float16)zv;
}

__global__ void log_softmax_kernel(float* __restrict__ logits, int N) {
  int row = blockIdx.x;
  float* p = logits + (size_t)row * N;
  int tid = threadIdx.x;  // 256
  __shared__ float r[256];
  float mx = -INFINITY;
  for (int i = tid; i < N; i += blockDim.x) mx = fmaxf(mx, p[i]);
  r[tid] = mx;
  __syncthreads();
  for (int o = 128; o > 0; o >>= 1) { if (tid < o) r[tid] = fmaxf(r[tid], r[tid + o]); __syncthreads(); }
  mx = r[0];
  __syncthreads();
  float s = 0.f;
  for (int i = tid; i < N; i += blockDim.x) s += __expf(p[i] - mx);
  r[tid] = s;
  __syncthreads();
  for (int o = 128; o > 0; o >>= 1) { if (tid < o) r[tid] += r[tid + o]; __syncthreads(); }
  float lse = __logf(r[0]) + mx;
  __syncthreads();
  for (int i = tid; i < N; i += blockDim.x) p[i] -= lse;
}

__global__ void zero_init_kernel(float* cE, _Float16* hE0, float* cD,
                                 _Float16* hD0, float* scal2) {
  int i = blockIdx.x * blockDim.x + threadIdx.x;
  if (i < 64 * 1024) {
    cE[i] = 0.f; hE0[i] = (_Float16)0.f;
    cD[i] = 0.f; hD0[i] = (_Float16)0.f;
  }
  if (i < 2) scal2[i] = 0.f;
}

// ---------------------------------------------------------------------------
// Host side
// ---------------------------------------------------------------------------
static inline char* bump_(char*& p, size_t bytes) {
  char* r = p;
  p += (bytes + 255) & ~(size_t)255;
  return r;
}

static inline void launch_gemm(const _Float16* A, const _Float16* W,
                               const float* b1, const float* b2,
                               float* C, _Float16* Ch,
                               int M, int N, int K, hipStream_t s) {
  int tiles = (M / 16) * (N / 16);
  int blocks = (tiles + 7) / 8;
  wmma_gemm_kernel<<<blocks, 256, 0, s>>>(A, W, b1, b2, C, Ch, M, N, K);
}

static inline void launch_cvt(const float* s, _Float16* d, long n, hipStream_t st) {
  cvt_f32_f16_kernel<<<(int)((n + 255) / 256), 256, 0, st>>>(s, d, n);
}

extern "C" void kernel_launch(void* const* d_in, const int* in_sizes, int n_in,
                              void* d_out, int out_size, void* d_ws, size_t ws_size,
                              hipStream_t stream) {
  const int Vv = 32000, E = 512, T = 64, H = 1024, Zz = 128, Bb = 64;
  const int BH = Bb * H;

  const int*   enc_input = (const int*)d_in[0];
  const int*   dec_input = (const int*)d_in[1];
  const int*   length    = (const int*)d_in[2];
  const float* eps       = (const float*)d_in[3];
  const float* enc_emb   = (const float*)d_in[4];
  const float* enc_Wih   = (const float*)d_in[5];
  const float* enc_Whh   = (const float*)d_in[6];
  const float* enc_bih   = (const float*)d_in[7];
  const float* enc_bhh   = (const float*)d_in[8];
  const float* latent_W  = (const float*)d_in[9];
  const float* latent_b  = (const float*)d_in[10];
  const float* dec_emb   = (const float*)d_in[11];
  const float* attn_W    = (const float*)d_in[12];
  const float* attn_b    = (const float*)d_in[13];
  const float* combine_W = (const float*)d_in[14];
  const float* combine_b = (const float*)d_in[15];
  const float* dec_Wih   = (const float*)d_in[16];
  const float* dec_Whh   = (const float*)d_in[17];
  const float* dec_bih   = (const float*)d_in[18];
  const float* dec_bhh   = (const float*)d_in[19];
  const float* pri_W     = (const float*)d_in[20];
  const float* pri_b     = (const float*)d_in[21];
  const float* inf_W     = (const float*)d_in[22];
  const float* inf_b     = (const float*)d_in[23];
  const float* aux_W     = (const float*)d_in[24];
  const float* aux_b     = (const float*)d_in[25];
  const float* out_W     = (const float*)d_in[26];
  const float* out_b     = (const float*)d_in[27];

  float* d_outF = (float*)d_out;
  const size_t MU_OFF = (size_t)Bb * (T + 1) * Vv;        // 133,120,000
  const size_t LV_OFF = MU_OFF + (size_t)Bb * Zz;
  const size_t SC_OFF = LV_OFF + (size_t)Bb * Zz;         // [kld_sum, aux_sum]

  // ---- workspace carve-up (f16 buffers then f32 buffers) ----
  char* p = (char*)d_ws;
  _Float16* encWih16 = (_Float16*)bump_(p, (size_t)4 * H * E * 2);
  _Float16* encWhh16 = (_Float16*)bump_(p, (size_t)4 * H * H * 2);
  _Float16* decWih16 = (_Float16*)bump_(p, (size_t)4 * H * H * 2);
  _Float16* decWhh16 = (_Float16*)bump_(p, (size_t)4 * H * H * 2);
  _Float16* latW16   = (_Float16*)bump_(p, (size_t)256 * 2048 * 2);
  _Float16* attnW16  = (_Float16*)bump_(p, (size_t)64 * 1536 * 2);
  _Float16* combW16  = (_Float16*)bump_(p, (size_t)1024 * 1792 * 2);
  _Float16* priW16   = (_Float16*)bump_(p, (size_t)256 * 1024 * 2);
  _Float16* infW16   = (_Float16*)bump_(p, (size_t)256 * 2048 * 2);
  _Float16* auxW16   = (_Float16*)bump_(p, (size_t)1024 * 128 * 2);
  _Float16* outW16   = (_Float16*)bump_(p, (size_t)Vv * H * 2);
  _Float16* xenc16   = (_Float16*)bump_(p, (size_t)T * Bb * E * 2);
  _Float16* xdec16   = (_Float16*)bump_(p, (size_t)(T + 1) * Bb * E * 2);
  _Float16* henc16   = (_Float16*)bump_(p, (size_t)2 * BH * 2);    // ping-pong
  _Float16* hdec16   = (_Float16*)bump_(p, (size_t)2 * BH * 2);    // ping-pong
  _Float16* hs16     = (_Float16*)bump_(p, (size_t)Bb * (T + 1) * H * 2);
  _Float16* cT16     = (_Float16*)bump_(p, (size_t)BH * 2);
  _Float16* ctx16    = (_Float16*)bump_(p, (size_t)BH * 2);
  _Float16* priMu16  = (_Float16*)bump_(p, (size_t)Bb * 128 * 2);
  _Float16* auxMu16  = (_Float16*)bump_(p, (size_t)BH * 2);
  _Float16* z16      = (_Float16*)bump_(p, (size_t)Bb * 128 * 2);
  _Float16* catXH    = (_Float16*)bump_(p, (size_t)Bb * 1536 * 2);
  _Float16* catHC    = (_Float16*)bump_(p, (size_t)Bb * 2048 * 2);
  _Float16* catHCtx  = (_Float16*)bump_(p, (size_t)Bb * 2048 * 2);
  _Float16* catComb  = (_Float16*)bump_(p, (size_t)Bb * 1792 * 2);
  _Float16* u16      = (_Float16*)bump_(p, (size_t)BH * 2);
  float* igates  = (float*)bump_(p, (size_t)T * Bb * 4 * H * 4);   // [T,B,4H]
  float* encOut  = (float*)bump_(p, (size_t)Bb * T * H * 4);       // [B,T,H]
  float* cEnc    = (float*)bump_(p, (size_t)BH * 4);
  float* cDec    = (float*)bump_(p, (size_t)BH * 4);
  float* lat     = (float*)bump_(p, (size_t)Bb * 256 * 4);
  float* pri     = (float*)bump_(p, (size_t)Bb * 256 * 4);
  float* infB    = (float*)bump_(p, (size_t)Bb * 256 * 4);
  float* scale   = (float*)bump_(p, (size_t)Bb * 64 * 4);
  float* ctx32   = (float*)bump_(p, (size_t)BH * 4);
  float* auxMu32 = (float*)bump_(p, (size_t)BH * 4);
  float* u32     = (float*)bump_(p, (size_t)BH * 4);

  // ---- init state + output scalars ----
  zero_init_kernel<<<256, 256, 0, stream>>>(cEnc, henc16, cDec, hdec16,
                                            d_outF + SC_OFF);

  // ---- weight conversions f32 -> f16 (one-time per call) ----
  launch_cvt(enc_Wih,   encWih16, (long)4 * H * E, stream);
  launch_cvt(enc_Whh,   encWhh16, (long)4 * H * H, stream);
  launch_cvt(dec_Wih,   decWih16, (long)4 * H * H, stream);
  launch_cvt(dec_Whh,   decWhh16, (long)4 * H * H, stream);
  launch_cvt(latent_W,  latW16,   (long)256 * 2048, stream);
  launch_cvt(attn_W,    attnW16,  (long)64 * 1536, stream);
  launch_cvt(combine_W, combW16,  (long)1024 * 1792, stream);
  launch_cvt(pri_W,     priW16,   (long)256 * 1024, stream);
  launch_cvt(inf_W,     infW16,   (long)256 * 2048, stream);
  launch_cvt(aux_W,     auxW16,   (long)1024 * 128, stream);
  launch_cvt(out_W,     outW16,   (long)Vv * H, stream);

  // ---- embeddings (time-major, f16) ----
  {
    long n1 = (long)T * Bb * E;
    gather_emb_kernel<<<(int)((n1 + 255) / 256), 256, 0, stream>>>(
        enc_emb, enc_input, xenc16, T, Bb, E);
    long n2 = (long)(T + 1) * Bb * E;
    gather_emb_kernel<<<(int)((n2 + 255) / 256), 256, 0, stream>>>(
        dec_emb, dec_input, xdec16, T + 1, Bb, E);
  }

  // ---- hoisted encoder input gates: [T*B, 4H] = x_enc @ Wih^T + bih + bhh ----
  launch_gemm(xenc16, encWih16, enc_bih, enc_bhh, igates, nullptr,
              T * Bb, 4 * H, E, stream);

  // ---- encoder recurrence (64 sequential fused WMMA+cell steps) ----
  for (int t = 0; t < T; ++t) {
    lstm_step_kernel<<<32, 256, 0, stream>>>(
        nullptr, 0, nullptr,
        igates + (size_t)t * Bb * 4 * H, 4 * H,
        henc16 + (size_t)(t & 1) * BH, encWhh16,
        nullptr, nullptr,
        cEnc,
        henc16 + (size_t)((t + 1) & 1) * BH,
        encOut + (size_t)t * H, T * H,
        nullptr, 0);
  }

  // ---- latent: z = mu + exp(0.5*logvar)*eps ----
  launch_cvt(cEnc, cT16, (long)BH, stream);
  concat4_kernel<<<(Bb * 2048 + 255) / 256, 256, 0, stream>>>(
      henc16 /* ping 0 after 64 steps */, H, cT16, H, nullptr, 0, nullptr, 0,
      catHC, Bb);
  launch_gemm(catHC, latW16, latent_b, nullptr, lat, nullptr, Bb, 256, 2048, stream);
  latent_z_kernel<<<32, 256, 0, stream>>>(lat, eps, d_outF + MU_OFF,
                                          d_outF + LV_OFF, z16);

  // ---- decoder recurrence (65 sequential steps) ----
  for (int t = 0; t < T + 1; ++t) {
    const _Float16* hcur  = hdec16 + (size_t)(t & 1) * BH;
    _Float16*       hnext = hdec16 + (size_t)((t + 1) & 1) * BH;
    const _Float16* xt    = xdec16 + (size_t)t * Bb * E;

    concat4_kernel<<<(Bb * 1536 + 255) / 256, 256, 0, stream>>>(
        xt, E, hcur, H, nullptr, 0, nullptr, 0, catXH, Bb);
    launch_gemm(catXH, attnW16, attn_b, nullptr, scale, nullptr, Bb, 64, 1536, stream);
    launch_gemm(hcur, priW16, pri_b, nullptr, pri, nullptr, Bb, 256, H, stream);
    attn_context_kernel<<<Bb, 256, 0, stream>>>(scale, enc_input, encOut,
                                                ctx32, ctx16);
    concat4_kernel<<<(Bb * 2048 + 255) / 256, 256, 0, stream>>>(
        hcur, H, ctx16, H, nullptr, 0, nullptr, 0, catHCtx, Bb);
    launch_gemm(catHCtx, infW16, inf_b, nullptr, infB, nullptr, Bb, 256, 2048, stream);
    kld_kernel<<<Bb, 128, 0, stream>>>(pri, infB, length, t, d_outF + SC_OFF);
    cvt_rows_kernel<<<(Bb * 128 + 255) / 256, 256, 0, stream>>>(pri, 256,
                                                                priMu16, 128, Bb);
    launch_gemm(priMu16, auxW16, aux_b, nullptr, auxMu32, auxMu16, Bb, H, 128, stream);
    auxc_kernel<<<Bb, 256, 0, stream>>>(ctx32, auxMu32, d_outF + SC_OFF + 1);
    concat4_kernel<<<(Bb * 1792 + 255) / 256, 256, 0, stream>>>(
        xt, E, auxMu16, H, priMu16, 128, z16, 128, catComb, Bb);
    launch_gemm(catComb, combW16, combine_b, nullptr, u32, u16, Bb, H, 1792, stream);
    lstm_step_kernel<<<32, 256, 0, stream>>>(
        u16, H, decWih16,
        nullptr, 0,
        hcur, decWhh16,
        dec_bih, dec_bhh,
        cDec,
        hnext,
        nullptr, 0,
        hs16 + (size_t)t * H, (T + 1) * H);
  }

  // ---- output projection [4160,1024] x [1024,32000]^T -> logits in d_out ----
  launch_gemm(hs16, outW16, out_b, nullptr, d_outF, nullptr,
              Bb * (T + 1), Vv, H, stream);

  // ---- in-place log_softmax over V per row ----
  log_softmax_kernel<<<Bb * (T + 1), 256, 0, stream>>>(d_outF, Vv);
}
